// InstEncoder_34110630265157
// MI455X (gfx1250) — matrix operand order repaired and checked by reading, not compiled, not run on hardware
//
#include <hip/hip_runtime.h>

// ---------------------------------------------------------------------------
// Shapes: N=8, I=J=32 -> M = N*I*J = 8192 rows; X=R=256, U=V=64, H=128.
// Workspace (ushort elems): t_tr 262144 | vrelu 1048576 | w2b 2097152 |
//                           rbf 2097152 | part (f32) 4*524288   (~18.5 MB)
// ---------------------------------------------------------------------------

typedef __bf16 v16bf __attribute__((ext_vector_type(16)));
typedef float  v8f   __attribute__((ext_vector_type(8)));

union Frag16 { v16bf v; unsigned short e[16]; unsigned d[8]; uint4 q[2]; };
union Acc    { v8f v; float f[8]; };

__device__ __forceinline__ unsigned short f2bfu(float f) {
  unsigned u = __builtin_bit_cast(unsigned, f);
  u += 0x7fffu + ((u >> 16) & 1u);            // round-to-nearest-even
  return (unsigned short)(u >> 16);
}

// packed bf16 multiply: lo*lo, hi*hi (CDNA5 VOP3P V_PK_MUL_BF16)
__device__ __forceinline__ unsigned pkmul_bf16(unsigned a, unsigned b) {
  unsigned d;
  asm("v_pk_mul_bf16 %0, %1, %2" : "=v"(d) : "v"(a), "v"(b));
  return d;
}

// Async DMA of a 16 KB block (global -> LDS), 128 threads x 8 x b128.
// GVS addressing: mem = SGPR64 + VGPR32 + INST_OFFSET; the same INST_OFFSET is
// added to the LDS address (ISA 08_async_tensor 4.4). Tracked by ASYNCcnt.
__device__ __forceinline__ void stage16k(unsigned short* dst,
                                         const unsigned short* src, int tid) {
  const unsigned lds  = (unsigned)(unsigned long long)dst + (unsigned)tid * 16u;
  const unsigned voff = (unsigned)tid * 16u;
  const unsigned long long sa = (unsigned long long)src;
  asm volatile(
      "global_load_async_to_lds_b128 %0, %1, %2 offset:0\n\t"
      "global_load_async_to_lds_b128 %0, %1, %2 offset:2048\n\t"
      "global_load_async_to_lds_b128 %0, %1, %2 offset:4096\n\t"
      "global_load_async_to_lds_b128 %0, %1, %2 offset:6144\n\t"
      "global_load_async_to_lds_b128 %0, %1, %2 offset:8192\n\t"
      "global_load_async_to_lds_b128 %0, %1, %2 offset:10240\n\t"
      "global_load_async_to_lds_b128 %0, %1, %2 offset:12288\n\t"
      "global_load_async_to_lds_b128 %0, %1, %2 offset:14336"
      :: "v"(lds), "v"(voff), "s"(sa)
      : "memory");
}

__device__ __forceinline__ void wait_async() {
  asm volatile("s_wait_asynccnt 0" ::: "memory");
}

// ---------------------------------------------------------------------------
// Kernel 1: t[n,r,h] = sum_u (sum_v u4[n,r,u,v]) * w1[r,u,h]
// stored transposed+bf16 as t_tr[n][h][r] for contiguous WMMA-B loads.
// ---------------------------------------------------------------------------
__global__ void k_t(const float* __restrict__ u4, const float* __restrict__ w1,
                    unsigned short* __restrict__ t_tr) {
  __shared__ float s_s[64];
  const int nr  = blockIdx.x;          // n*256 + r
  const int n   = nr >> 8;
  const int rr  = nr & 255;
  const int tid = threadIdx.x;         // 0..127
  const float* up = u4 + (size_t)nr * 64 * 64;
  if (tid < 64) {
    float s = 0.f;
    const float* p = up + tid * 64;
#pragma unroll 8
    for (int v = 0; v < 64; ++v) s += p[v];
    s_s[tid] = s;
  }
  __syncthreads();
  const int h = tid;
  const float* w1p = w1 + (size_t)rr * 64 * 128 + h;
  float t = 0.f;
#pragma unroll 8
  for (int uu = 0; uu < 64; ++uu) t += s_s[uu] * w1p[uu * 128];
  t_tr[((size_t)n * 128 + h) * 256 + rr] = f2bfu(t);
}

// ---------------------------------------------------------------------------
// Kernel 2: element-wise f32 -> bf16 convert (used for w2 and r; both 2097152).
// ---------------------------------------------------------------------------
__global__ void k_cvt(const float* __restrict__ src, unsigned short* __restrict__ dst) {
  const int i = blockIdx.x * 256 + threadIdx.x;
  dst[i] = f2bfu(src[i]);
}

// ---------------------------------------------------------------------------
// Kernel 3: vrelu[m,h] = relu( sum_k r[m,k] * t[n(m),k,h] ), bf16 WMMA.
// ---------------------------------------------------------------------------
__global__ void __launch_bounds__(128)
k_vrelu(const unsigned short* __restrict__ rbf,
        const unsigned short* __restrict__ t_tr,
        unsigned short* __restrict__ vrelu) {
  const int tile = blockIdx.x;              // 0..511
  const int wv   = threadIdx.x >> 5;        // 0..3
  const int lane = threadIdx.x & 31;
  const int m    = lane & 15;
  const int hlf  = lane >> 4;
  const int mrow = tile * 16 + m;
  const int n    = tile >> 6;               // batch (1024 rows per batch)
  const int h0   = wv * 32;
  Acc acc0, acc1;
#pragma unroll
  for (int g = 0; g < 8; ++g) { acc0.f[g] = 0.f; acc1.f[g] = 0.f; }
  const unsigned short* rp = rbf + (size_t)mrow * 256;

  for (int kc = 0; kc < 8; ++kc) {
    const int k0 = kc * 32;
    // A fragment (16x32 bf16): lanes 0-15 K=0..7,16..23; lanes 16-31 K=8..15,24..31
    Frag16 a;
    a.q[0] = *(const uint4*)(rp + k0 + hlf * 8);
    a.q[1] = *(const uint4*)(rp + k0 + 16 + hlf * 8);
#pragma unroll
    for (int bi = 0; bi < 2; ++bi) {
      const int hcol = h0 + bi * 16 + m;
      const unsigned short* bp =
          t_tr + ((size_t)n * 128 + hcol) * 256 + k0 + hlf * 8;
      Frag16 b;
      b.q[0] = ((const uint4*)bp)[0];
      b.q[1] = ((const uint4*)bp)[2];
      Acc& acc = bi ? acc1 : acc0;
      acc.v = __builtin_amdgcn_wmma_f32_16x16x32_bf16(
          false, a.v, false, b.v, (short)0, acc.v, false, false);
    }
  }
#pragma unroll
  for (int bi = 0; bi < 2; ++bi) {
    Acc& acc = bi ? acc1 : acc0;
#pragma unroll
    for (int g = 0; g < 8; ++g) {
      const int row = tile * 16 + g + hlf * 8;
      const int h   = h0 + bi * 16 + m;
      const float vv = acc.f[g];
      vrelu[(size_t)row * 128 + h] = f2bfu(vv > 0.f ? vv : 0.f);
    }
  }
}

// ---------------------------------------------------------------------------
// Kernel 4 (dominant, 34.4 GFLOP): out[m,v] = sum_{x,h} x[m,x]*vrelu[m,h]*w2[x,v,h]
// A[m,(x,h)] = x[m,x]*vrelu[m,h] built with v_pk_mul_bf16;
// B[(x,h),v] = w2b[x][v][h], double-buffered into LDS with async-to-LDS DMA.
// Split-K over x (4 ways) into f32 partials.
// ---------------------------------------------------------------------------
__global__ void __launch_bounds__(128)
k_gemm(const float* __restrict__ x, const unsigned short* __restrict__ vrelu,
       const unsigned short* __restrict__ w2b, float* __restrict__ part) {
  __shared__ __align__(16) unsigned short s_vr[4][16][128];  // 16 KB
  __shared__ __align__(16) float          s_x [4][16][64];   // 16 KB
  __shared__ __align__(16) unsigned short s_b [2][8192];     // 32 KB (double buf)
  const int tid  = threadIdx.x;
  const int wv   = tid >> 5;
  const int lane = tid & 31;
  const int tile = blockIdx.x * 4 + wv;         // 0..511
  const int xs   = blockIdx.y;                  // 0..3 split-K over x
  const int m0   = tile * 16;

  // Stage vrelu 16x64 bf16 (h 0..63 of each row) into LDS (4 x uint4 per lane)
  for (int t = lane; t < 128; t += 32) {
    const int row = t >> 3, c = t & 7;
    ((uint4*)&s_vr[wv][row][0])[c] =
        ((const uint4*)(vrelu + (size_t)(m0 + row) * 128))[c];
  }
  // Stage x 16x64 f32 tile into LDS (8 x float4 per lane)
  for (int t = lane; t < 256; t += 32) {
    const int row = t >> 4, c = t & 15;
    ((float4*)&s_x[wv][row][0])[c] =
        ((const float4*)(x + (size_t)(m0 + row) * 256 + xs * 64))[c];
  }
  // Kick async DMA of the first B x-slice (16 KB) into buffer 0
  stage16k(&s_b[0][0], w2b + (size_t)(xs * 64) * 8192, tid);
  wait_async();
  __syncthreads();

  const int m = lane & 15, hlf = lane >> 4;
  Acc acc[4];
#pragma unroll
  for (int vt = 0; vt < 4; ++vt)
#pragma unroll
    for (int g = 0; g < 8; ++g) acc[vt].f[g] = 0.f;

  for (int xi = 0; xi < 64; ++xi) {
    const int cur = xi & 1;
    // overlap: DMA next x-slice while computing on the current one
    if (xi < 63)
      stage16k(&s_b[cur ^ 1][0], w2b + (size_t)(xs * 64 + xi + 1) * 8192, tid);

    const unsigned short* bb = &s_b[cur][0];
    // broadcast x[m, xcur] as a packed bf16 pair
    const unsigned xpair = (unsigned)f2bfu(s_x[wv][m][xi]) * 0x10001u;
#pragma unroll
    for (int kc = 0; kc < 4; ++kc) {
      const int h0 = kc * 32;
      // A fragment: a[m,k] = x[m,xcur] * vrelu[m, h0 + K(i,lane)], pairs in dwords
      const uint4 va = *(const uint4*)&s_vr[wv][m][h0 + hlf * 8];
      const uint4 vb = *(const uint4*)&s_vr[wv][m][h0 + 16 + hlf * 8];
      Frag16 a;
      a.d[0] = pkmul_bf16(xpair, va.x);
      a.d[1] = pkmul_bf16(xpair, va.y);
      a.d[2] = pkmul_bf16(xpair, va.z);
      a.d[3] = pkmul_bf16(xpair, va.w);
      a.d[4] = pkmul_bf16(xpair, vb.x);
      a.d[5] = pkmul_bf16(xpair, vb.y);
      a.d[6] = pkmul_bf16(xpair, vb.z);
      a.d[7] = pkmul_bf16(xpair, vb.w);
#pragma unroll
      for (int vt = 0; vt < 4; ++vt) {
        const unsigned short* bp = bb + (vt * 16 + m) * 128 + h0 + hlf * 8;
        Frag16 b;
        b.q[0] = ((const uint4*)bp)[0];
        b.q[1] = ((const uint4*)bp)[2];
        acc[vt].v = __builtin_amdgcn_wmma_f32_16x16x32_bf16(
            false, a.v, false, b.v, (short)0, acc[vt].v, false, false);
      }
    }
    // next buffer ready + nobody still reading the buffer we'll overwrite
    wait_async();
    __syncthreads();
  }

  float* pp = part + (size_t)xs * (8192 * 64);
#pragma unroll
  for (int vt = 0; vt < 4; ++vt)
#pragma unroll
    for (int g = 0; g < 8; ++g)
      pp[(size_t)(m0 + g + hlf * 8) * 64 + vt * 16 + m] = acc[vt].f[g];
}

// ---------------------------------------------------------------------------
// Kernel 5: reduce 4 split-K partials -> out (f32).
// ---------------------------------------------------------------------------
__global__ void k_reduce(const float* __restrict__ part, float* __restrict__ out) {
  const int i = blockIdx.x * 256 + threadIdx.x;   // 524288 total
  out[i] = part[i] + part[i + 524288] + part[i + 2 * 524288] + part[i + 3 * 524288];
}

// ---------------------------------------------------------------------------
extern "C" void kernel_launch(void* const* d_in, const int* in_sizes, int n_in,
                              void* d_out, int out_size, void* d_ws, size_t ws_size,
                              hipStream_t stream) {
  (void)in_sizes; (void)n_in; (void)out_size; (void)ws_size;
  const float* x  = (const float*)d_in[0];   // (8,32,32,256)
  const float* r  = (const float*)d_in[1];   // (8,32,32,256)
  const float* u4 = (const float*)d_in[2];   // (8,256,64,64)
  const float* w1 = (const float*)d_in[3];   // (256,64,128)
  const float* w2 = (const float*)d_in[4];   // (256,64,128)
  float* out = (float*)d_out;                // (8,32,32,64)

  unsigned short* t_tr  = (unsigned short*)d_ws;     // 262144 bf16
  unsigned short* vrelu = t_tr + 262144;             // 1048576 bf16
  unsigned short* w2b   = vrelu + 1048576;           // 2097152 bf16
  unsigned short* rbf   = w2b + 2097152;             // 2097152 bf16
  float*          part  = (float*)(rbf + 2097152);   // 4 * 524288 f32

  k_t     <<<2048, 128, 0, stream>>>(u4, w1, t_tr);
  k_cvt   <<<8192, 256, 0, stream>>>(w2, w2b);
  k_cvt   <<<8192, 256, 0, stream>>>(r, rbf);
  k_vrelu <<<512,  128, 0, stream>>>(rbf, t_tr, vrelu);
  k_gemm  <<<dim3(128, 4), 128, 0, stream>>>(x, vrelu, w2b, part);
  k_reduce<<<2048, 256, 0, stream>>>(part, out);
}
